// ScaledDotProductAttention_4698694222223
// MI455X (gfx1250) — matrix operand order repaired and checked by reading, not compiled, not run on hardware
//
#include <hip/hip_runtime.h>
#include <cstdint>

#define BHN   64      // B*H
#define LSEQ  2048
#define DHEAD 64

typedef __attribute__((ext_vector_type(16))) __bf16 v16bf;
typedef __attribute__((ext_vector_type(8)))  float  v8f;
typedef __attribute__((ext_vector_type(4)))  float  f4v;

union BfOp { v16bf v; f4v q[2]; };

__device__ __forceinline__ v8f wmma_bf16(v16bf a, v16bf b, v8f c) {
  // D = A(16x32 bf16) * B(32x16 bf16) + C(16x16 f32)
  return __builtin_amdgcn_wmma_f32_16x16x32_bf16(
      /*neg_a=*/false, a, /*neg_b=*/false, b,
      /*c_mod=*/(short)0, c, /*reuse_a=*/false, /*reuse_b=*/false);
}

// A operand: row = lane%16 (contiguous K in memory). lanes<16 hold K 0..15
// of the 32-chunk, lanes>=16 hold K 16..31. off = chunk + sel*16.
__device__ __forceinline__ v16bf load_tileA(const __bf16* rowbase, int off) {
  BfOp t;
  t.q[0] = *reinterpret_cast<const f4v*>(rowbase + off);
  t.q[1] = *reinterpret_cast<const f4v*>(rowbase + off + 8);
  return t.v;
}
// B operand: col = lane%16, 16 contiguous K values starting at off.
__device__ __forceinline__ v16bf load_tileB(const __bf16* rowbase, int off) {
  BfOp t;
  t.q[0] = *reinterpret_cast<const f4v*>(rowbase + off);
  t.q[1] = *reinterpret_cast<const f4v*>(rowbase + off + 8);
  return t.v;
}

__device__ __forceinline__ __bf16 f2bf(float f) { return (__bf16)f; }

// Intra-wave LDS fence: DS ops from one wave are processed in order by the
// LDS pipe (ISA 7.3), so draining DScnt makes this wave's prior ds_stores
// visible to its own subsequent ds_loads. "memory" stops IR reordering.
__device__ __forceinline__ void wave_lds_fence() {
  asm volatile("s_wait_dscnt 0x0" ::: "memory");
}

// ---------------------------------------------------------------------------
// Prep: f32 [BH,64,L] -> bf16 transposed [BH,L,64] via LDS tile (both sides
// coalesced; LDS row stride 66*2B = 33 dwords -> conflict-free read).
// ---------------------------------------------------------------------------
__global__ __launch_bounds__(256)
void transpose_to_bf16(const float* __restrict__ src, __bf16* __restrict__ dst) {
  __shared__ __bf16 tile[64][66];
  const int bh = blockIdx.x;
  const int l0 = blockIdx.y * 64;
  const float* s = src + (size_t)bh * DHEAD * LSEQ;
  __bf16*      d = dst + (size_t)bh * LSEQ * DHEAD;
  #pragma unroll
  for (int i = 0; i < 16; ++i) {
    int idx = i * 256 + threadIdx.x;
    int dd = idx >> 6, ll = idx & 63;
    tile[dd][ll] = f2bf(s[(size_t)dd * LSEQ + l0 + ll]);
  }
  __syncthreads();
  #pragma unroll
  for (int i = 0; i < 16; ++i) {
    int idx = i * 256 + threadIdx.x;
    int ll = idx >> 6, dd = idx & 63;
    d[(size_t)(l0 + ll) * DHEAD + dd] = tile[dd][ll];
  }
}

// Prep: f32 -> bf16 elementwise (V keeps its [BH,64,L] layout).
__global__ __launch_bounds__(256)
void convert_v_bf16(const float4* __restrict__ src, __bf16* __restrict__ dst, int n4) {
  int i = blockIdx.x * 256 + threadIdx.x;
  if (i < n4) {
    float4 f = src[i];
    union { __bf16 h[4]; uint64_t u; } t;
    t.h[0] = f2bf(f.x); t.h[1] = f2bf(f.y); t.h[2] = f2bf(f.z); t.h[3] = f2bf(f.w);
    *reinterpret_cast<uint64_t*>(dst + (size_t)i * 4) = t.u;
  }
}

// ---------------------------------------------------------------------------
// Phase 1: per-q softmax stats. One wave owns a 16-q tile, streams all k.
// S^T tile = (K^T tile as A) x (Q as B): C[M=k][N=q]; each lane owns one q
// column and 8 k values per tile -> online max/sum, final merge lane^16.
// ---------------------------------------------------------------------------
__global__ __launch_bounds__(256)
void attn_rowstats(const __bf16* __restrict__ QT, const __bf16* __restrict__ KT,
                   float* __restrict__ mrow, float* __restrict__ lrow) {
  const int lane = threadIdx.x & 31;
  const int wave = threadIdx.x >> 5;
  const int bh = blockIdx.x;
  const int q0 = (blockIdx.y * 8 + wave) * 16;
  const int r = lane & 15, sel = lane >> 4;
  const __bf16* qtb = QT + (size_t)bh * LSEQ * DHEAD;
  const __bf16* ktb = KT + (size_t)bh * LSEQ * DHEAD;

  // B operand (Q): fixed for the whole k-loop. Two 32-deep d-chunks.
  const __bf16* qrow = qtb + (size_t)(q0 + r) * DHEAD;
  const v16bf bq0 = load_tileB(qrow, sel * 16);        // d 0..31
  const v16bf bq1 = load_tileB(qrow, 32 + sel * 16);   // d 32..63

  const float scale = 0.125f;  // 1/temperature
  float mr = -INFINITY, lr = 0.f;

  for (int k0 = 0; k0 < LSEQ; k0 += 16) {
    const __bf16* krow = ktb + (size_t)(k0 + r) * DHEAD;
    v16bf a0 = load_tileA(krow, sel * 16);             // d 0..31
    v16bf a1 = load_tileA(krow, 32 + sel * 16);        // d 32..63
    v8f c = {};
    c = wmma_bf16(a0, bq0, c);
    c = wmma_bf16(a1, bq1, c);
    float tm = mr;
    #pragma unroll
    for (int j = 0; j < 8; ++j) tm = fmaxf(tm, c[j] * scale);
    float ls = 0.f;
    #pragma unroll
    for (int j = 0; j < 8; ++j) ls += __expf(c[j] * scale - tm);
    lr = lr * __expf(mr - tm) + ls;
    mr = tm;
  }
  // lanes n and n^16 hold the same q, disjoint k halves -> merge.
  float mo = __shfl_xor(mr, 16, 32);
  float lo = __shfl_xor(lr, 16, 32);
  float mm = fmaxf(mr, mo);
  float ll = lr * __expf(mr - mm) + lo * __expf(mo - mm);
  if (lane < 16) {
    mrow[(size_t)bh * LSEQ + q0 + lane] = mm;
    lrow[(size_t)bh * LSEQ + q0 + lane] = ll;
  }
}

// ---------------------------------------------------------------------------
// Phase 2: O^T = P^T V^T. One wave owns a 16-k output tile [16k x 64d],
// loops over q in chunks of 32: recompute two S^T tiles, form P^T in a
// wave-private LDS tile (C-layout -> A-layout relayout, intra-wave DS fence
// only -- no inter-wave barrier), then 4 WMMAs against V (B operand,
// contiguous q in the original [64,L] layout).
// ---------------------------------------------------------------------------
__global__ __launch_bounds__(256)
void attn_out(const __bf16* __restrict__ QT, const __bf16* __restrict__ KT,
              const __bf16* __restrict__ Vb, const float* __restrict__ mrow,
              const float* __restrict__ lrow, float* __restrict__ out) {
  const int lane = threadIdx.x & 31;
  const int wave = threadIdx.x >> 5;
  const int bh = blockIdx.x;
  const int k0 = (blockIdx.y * 8 + wave) * 16;
  const int r = lane & 15, sel = lane >> 4;

  // Per-wave P^T staging tile: 16 rows (k) x 32 cols (q) bf16.
  // Row stride 72 elems = 144 B: 16B-aligned b128 rows, 36-dword stride.
  __shared__ __bf16 plds_all[8][16 * 72];
  __bf16* plds = plds_all[wave];

  const __bf16* qtb = QT + (size_t)bh * LSEQ * DHEAD;
  const __bf16* ktb = KT + (size_t)bh * LSEQ * DHEAD;
  const __bf16* vb  = Vb + (size_t)bh * DHEAD * LSEQ;
  const float*  mb  = mrow + (size_t)bh * LSEQ;
  const float*  lb  = lrow + (size_t)bh * LSEQ;

  // A operand (K^T) is fixed: this wave's 16 k rows.
  const __bf16* krow = ktb + (size_t)(k0 + r) * DHEAD;
  const v16bf ak0 = load_tileA(krow, sel * 16);
  const v16bf ak1 = load_tileA(krow, 32 + sel * 16);

  // V row pointers per 16-d output tile (col d = dt*16 + r, contiguous q).
  const __bf16* vrow0 = vb + (size_t)(0 * 16 + r) * LSEQ;
  const __bf16* vrow1 = vb + (size_t)(1 * 16 + r) * LSEQ;
  const __bf16* vrow2 = vb + (size_t)(2 * 16 + r) * LSEQ;
  const __bf16* vrow3 = vb + (size_t)(3 * 16 + r) * LSEQ;

  v8f acc[4] = {};
  const float scale = 0.125f;

  for (int q0 = 0; q0 < LSEQ; q0 += 32) {
    #pragma unroll
    for (int t = 0; t < 2; ++t) {
      const int qq = q0 + t * 16;
      const __bf16* qrow = qtb + (size_t)(qq + r) * DHEAD;
      v16bf b0 = load_tileB(qrow, sel * 16);
      v16bf b1 = load_tileB(qrow, 32 + sel * 16);
      v8f cs = {};
      cs = wmma_bf16(ak0, b0, cs);
      cs = wmma_bf16(ak1, b1, cs);
      // This lane's column q = qq + r; rows k_local = j + 8*sel.
      const float mq = mb[qq + r];
      const float inv_l = 1.0f / lb[qq + r];
      #pragma unroll
      for (int j = 0; j < 8; ++j) {
        float p = __expf(cs[j] * scale - mq) * inv_l;
        plds[(j + 8 * sel) * 72 + t * 16 + r] = f2bf(p);
      }
    }
    // Staging tile is wave-private: an intra-wave DS drain orders the
    // cross-lane store->load handoff; DS in-order per wave covers the
    // load->next-store WAR at the LDS pipe.
    wave_lds_fence();
    // Reload P^T in A-operand layout: row k = r, q chunks per lane half.
    BfOp pa;
    pa.q[0] = *reinterpret_cast<const f4v*>(plds + r * 72 + sel * 16);
    pa.q[1] = *reinterpret_cast<const f4v*>(plds + r * 72 + 8 + sel * 16);
    // O^T tiles: acc[dt] += P^T(16k x 32q) * V^T(32q x 16d)
    acc[0] = wmma_bf16(pa.v, load_tileB(vrow0, q0 + sel * 16), acc[0]);
    acc[1] = wmma_bf16(pa.v, load_tileB(vrow1, q0 + sel * 16), acc[1]);
    acc[2] = wmma_bf16(pa.v, load_tileB(vrow2, q0 + sel * 16), acc[2]);
    acc[3] = wmma_bf16(pa.v, load_tileB(vrow3, q0 + sel * 16), acc[3]);
    // Compiler-only ordering so next-iteration stores are not hoisted above
    // the ds_loads; hardware keeps same-wave DS ops in order.
    asm volatile("" ::: "memory");
  }

  // Store: C element (M=j+8*sel -> k, N=r + 16*dt -> d); out[d*L + k],
  // 8 contiguous k per lane -> two float4 stores per d-tile.
  float* ob = out + (size_t)bh * DHEAD * LSEQ;
  #pragma unroll
  for (int dt = 0; dt < 4; ++dt) {
    float* op = ob + (size_t)(dt * 16 + r) * LSEQ + k0 + sel * 8;
    f4v lo = {acc[dt][0], acc[dt][1], acc[dt][2], acc[dt][3]};
    f4v hi = {acc[dt][4], acc[dt][5], acc[dt][6], acc[dt][7]};
    *reinterpret_cast<f4v*>(op) = lo;
    *reinterpret_cast<f4v*>(op + 4) = hi;
  }
}

// ---------------------------------------------------------------------------
extern "C" void kernel_launch(void* const* d_in, const int* in_sizes, int n_in,
                              void* d_out, int out_size, void* d_ws, size_t ws_size,
                              hipStream_t stream) {
  (void)in_sizes; (void)n_in; (void)out_size; (void)ws_size;
  const float* q = (const float*)d_in[0];
  const float* k = (const float*)d_in[1];
  const float* v = (const float*)d_in[2];
  float* out = (float*)d_out;

  const size_t nElems = (size_t)BHN * LSEQ * DHEAD;  // 8,388,608
  __bf16* QT = (__bf16*)d_ws;                        // [BH, L, 64] bf16 (16 MB)
  __bf16* KT = QT + nElems;                          // [BH, L, 64] bf16 (16 MB)
  __bf16* Vb = KT + nElems;                          // [BH, 64, L] bf16 (16 MB)
  float* mrow = (float*)(Vb + nElems);               // [BH, L] f32
  float* lrow = mrow + (size_t)BHN * LSEQ;           // [BH, L] f32

  transpose_to_bf16<<<dim3(BHN, LSEQ / 64), 256, 0, stream>>>(q, QT);
  transpose_to_bf16<<<dim3(BHN, LSEQ / 64), 256, 0, stream>>>(k, KT);
  const int n4 = (int)(nElems / 4);
  convert_v_bf16<<<(n4 + 255) / 256, 256, 0, stream>>>((const float4*)v, Vb, n4);

  attn_rowstats<<<dim3(BHN, LSEQ / 16 / 8), 256, 0, stream>>>(QT, KT, mrow, lrow);
  attn_out<<<dim3(BHN, LSEQ / 16 / 8), 256, 0, stream>>>(QT, KT, Vb, mrow, lrow, out);
}
